// Learner_Matcher_78838419686004
// MI455X (gfx1250) — compile-verified, matrix operand order checked by reading, not run on hardware
//
#include <hip/hip_runtime.h>
#include <hip/hip_bf16.h>
#include <hip/hip_fp16.h>

// ---------------------------------------------------------------------------
// Problem constants (from reference)
// ---------------------------------------------------------------------------
#define ENC    128
#define HID    256
#define GATE3  768           // 3*HID
#define BATCH  64
#define TLEN   128
#define VOCABN 50000
#define RROOTS 8192

typedef __attribute__((ext_vector_type(16))) _Float16 v16h;
typedef __attribute__((ext_vector_type(8)))  _Float16 v8h;
typedef __attribute__((ext_vector_type(8)))  float    v8f;

// ---------------------------------------------------------------------------
// WMMA A-style fragment loader for a 16x32 (rows x K) f16 tile, row-major,
// leading dimension ld.  Per CDNA5 ISA 16-bit A layout:
//   lane L: row = L&15, half = L>>4
//   VGPR 0..3 hold k = half*8 + {0..7}        (two f16 per VGPR)
//   VGPR 4..7 hold k = 16 + half*8 + {0..7}
// -> two contiguous 16-byte loads per lane.
// Used for A (M x K) and for B supplied as N x K row-major (i.e. B^T),
// so C = A * B^T which matches x @ W.T in the reference.
// ---------------------------------------------------------------------------
__device__ __forceinline__ v16h load_frag_nk(const _Float16* base, int ld) {
  const int lane = threadIdx.x & 31;
  const int row  = lane & 15;
  const int half = lane >> 4;
  const _Float16* p = base + (size_t)row * ld + half * 8;
  v8h lo = *(const v8h*)(p);
  v8h hi = *(const v8h*)(p + 16);
  v16h out;
#pragma unroll
  for (int i = 0; i < 8; ++i) { out[i] = lo[i]; out[8 + i] = hi[i]; }
  return out;
}

// ---------------------------------------------------------------------------
// Elementwise helpers
// ---------------------------------------------------------------------------
__global__ void k_zero_f32(float* __restrict__ p, int n) {
  int i = blockIdx.x * blockDim.x + threadIdx.x;
  if (i < n) p[i] = 0.0f;
}

__global__ void k_f32_to_f16(const float* __restrict__ s, _Float16* __restrict__ d, int n) {
  int i = blockIdx.x * blockDim.x + threadIdx.x;
  if (i < n) d[i] = (_Float16)s[i];
}

// ---------------------------------------------------------------------------
// C(MxN,f32) = A(MxK,f16 rowmajor) * B(NxK,f16 rowmajor)^T + bias(N)
// One wave per 16x16 output tile.  M,N multiples of 16; K multiple of 32.
// ---------------------------------------------------------------------------
__global__ void __launch_bounds__(128)
k_gemm_nt(const _Float16* __restrict__ A, const _Float16* __restrict__ B,
          const float* __restrict__ bias, float* __restrict__ C,
          int M, int N, int K) {
  const int wave    = blockIdx.x * (blockDim.x >> 5) + (threadIdx.x >> 5);
  const int tiles_n = N >> 4;
  const int tiles_m = M >> 4;
  if (wave >= tiles_m * tiles_n) return;   // wave-uniform exit: EXEC stays full
  const int tm = wave / tiles_n;
  const int tn = wave - tm * tiles_n;

  const _Float16* a0 = A + (size_t)tm * 16 * K;
  const _Float16* b0 = B + (size_t)tn * 16 * K;

  v8f acc = {};
  for (int k = 0; k < K; k += 32) {
    v16h af = load_frag_nk(a0 + k, K);
    v16h bf = load_frag_nk(b0 + k, K);
    acc = __builtin_amdgcn_wmma_f32_16x16x32_f16(false, af, false, bf,
                                                 (short)0, acc, false, false);
  }

  const int lane = threadIdx.x & 31;
  const int col  = lane & 15;
  const int half = lane >> 4;
  const int gcol = tn * 16 + col;
  const float bv = bias ? bias[gcol] : 0.0f;
#pragma unroll
  for (int v = 0; v < 8; ++v) {
    int gm = tm * 16 + v + 8 * half;
    C[(size_t)gm * N + gcol] = acc[v] + bv;
  }
}

// ---------------------------------------------------------------------------
// Tree phase kernels.  cur/prev are (count x 128) f32.
// ---------------------------------------------------------------------------
__global__ void k_gather_proj(const float* __restrict__ P, const int* __restrict__ tok,
                              float* __restrict__ cur, int count) {
  long i = (long)blockIdx.x * blockDim.x + threadIdx.x;
  if (i >= (long)count * ENC) return;
  int node = (int)(i >> 7), e = (int)(i & 127);
  cur[i] = P[(size_t)tok[node] * ENC + e];
}

__global__ void k_scatter_add(float* __restrict__ cur, const float* __restrict__ child,
                              const int* __restrict__ pidx, int count) {
  long i = (long)blockIdx.x * blockDim.x + threadIdx.x;
  if (i >= (long)count * ENC) return;
  int node = (int)(i >> 7), e = (int)(i & 127);
  atomicAdd(&cur[(size_t)pidx[node] * ENC + e], child[i]);
}

// segment_max fused with max(.,0): roots pre-zeroed; only v>0 can matter and
// int-compare of positive float bits is monotonic -> exact & deterministic.
__global__ void k_pool_max(const float* __restrict__ cur, const int* __restrict__ bidx,
                           int* __restrict__ roots_bits, int count) {
  long i = (long)blockIdx.x * blockDim.x + threadIdx.x;
  if (i >= (long)count * ENC) return;
  int node = (int)(i >> 7), e = (int)(i & 127);
  float v = cur[i];
  if (v > 0.0f)
    atomicMax(&roots_bits[(size_t)bidx[node] * ENC + e], __float_as_int(v));
}

// ---------------------------------------------------------------------------
// GRU scan.  grid.x = 2 (dir 0 fwd, dir 1 bwd), 1024 threads = 32 waves.
// gi: (B*T, 768) f32 precomputed x@wih^T+bih.  whh: (768,256) f16.
// LDS:  gh (64x768 f32, 192KB) + h (64x256 f16, 32KB)  = 229376 B  (<=320KB/WG)
// Per step: gh = h @ whh^T + bhh via 192 WMMA tiles (6/wave, K=256 -> 8 wmma),
// then fused gate math, h and y update.  Only block barriers needed.
// ---------------------------------------------------------------------------
#define SCAN_LDS_BYTES (BATCH * GATE3 * 4 + BATCH * HID * 2)

__global__ void __launch_bounds__(1024)
k_gru_scan(const float* __restrict__ giF, const float* __restrict__ giB,
           const _Float16* __restrict__ whhF, const _Float16* __restrict__ whhB,
           const float* __restrict__ bhhF, const float* __restrict__ bhhB,
           float* __restrict__ y) {
  extern __shared__ unsigned char smem[];
  float*    gh  = (float*)smem;                                   // 64 x 768
  _Float16* hsh = (_Float16*)(smem + (size_t)BATCH * GATE3 * 4);  // 64 x 256

  const int dir = blockIdx.x;
  const float*    gi  = dir ? giB  : giF;
  const _Float16* whh = dir ? whhB : whhF;
  const float*    bhh = dir ? bhhB : bhhF;

  const int tid  = threadIdx.x;
  const int wv   = tid >> 5;
  const int lane = tid & 31;
  const int col  = lane & 15;
  const int half = lane >> 4;

  for (int i = tid; i < BATCH * HID; i += 1024) hsh[i] = (_Float16)0.0f;
  __syncthreads();

  for (int t = 0; t < TLEN; ++t) {
    const int ta = dir ? (TLEN - 1 - t) : t;

    // ---- gh = h @ whh^T + bhh -------------------------------------------
#pragma unroll
    for (int i = 0; i < 6; ++i) {
      const int tile = wv * 6 + i;     // 0..191
      const int tm = tile / 48;        // 0..3   (M tiles of 64)
      const int tn = tile % 48;        // 0..47  (N tiles of 768)
      const _Float16* a0 = hsh + (size_t)tm * 16 * HID;
      const _Float16* b0 = whh + (size_t)tn * 16 * HID;
      v8f acc = {};
#pragma unroll
      for (int k = 0; k < HID; k += 32) {
        v16h af = load_frag_nk(a0 + k, HID);   // LDS -> ds_load_b128
        v16h bf = load_frag_nk(b0 + k, HID);   // global (L2 resident)
        acc = __builtin_amdgcn_wmma_f32_16x16x32_f16(false, af, false, bf,
                                                     (short)0, acc, false, false);
      }
      const int gcol = tn * 16 + col;
      const float bv = bhh[gcol];
#pragma unroll
      for (int v = 0; v < 8; ++v) {
        int gm = tm * 16 + v + 8 * half;
        gh[gm * GATE3 + gcol] = acc[v] + bv;
      }
    }
    __syncthreads();

    // ---- gates + state update (each (b,j) owned by exactly one thread) --
#pragma unroll
    for (int i = 0; i < (BATCH * HID) / 1024; ++i) {
      const int idx = tid + i * 1024;
      const int b = idx >> 8;
      const int j = idx & 255;
      const float* gr = gi + (size_t)(b * TLEN + ta) * GATE3;
      const float ir  = gr[j];
      const float iz  = gr[HID + j];
      const float inn = gr[2 * HID + j];
      const float hr  = gh[b * GATE3 + j];
      const float hz  = gh[b * GATE3 + HID + j];
      const float hn  = gh[b * GATE3 + 2 * HID + j];
      const float r = 1.0f / (1.0f + __expf(-(ir + hr)));
      const float z = 1.0f / (1.0f + __expf(-(iz + hz)));
      const float n = tanhf(inn + r * hn);
      const float hold = (float)hsh[b * HID + j];
      const float hnew = (1.0f - z) * n + z * hold;
      y[(size_t)(b * TLEN + ta) * (2 * HID) + dir * HID + j] = hnew;
      hsh[b * HID + j] = (_Float16)hnew;
    }
    __syncthreads();
  }
}

// out[b, c] = max_t y[b, t, c]   (c in [0,512))
__global__ void k_time_max(const float* __restrict__ y, float* __restrict__ out) {
  int g = blockIdx.x * blockDim.x + threadIdx.x;
  if (g >= BATCH * 2 * HID) return;
  int b = g >> 9, c = g & 511;
  const float* p = y + (size_t)b * TLEN * (2 * HID) + c;
  float m = p[0];
  for (int t = 1; t < TLEN; ++t) m = fmaxf(m, p[(size_t)t * (2 * HID)]);
  out[g] = m;
}

// ---------------------------------------------------------------------------
// Host orchestration
// ---------------------------------------------------------------------------
static inline int grid1d(long n, int b) { return (int)((n + b - 1) / b); }
static inline int gemm_grid(int M, int N) { return ((M / 16) * (N / 16) + 3) / 4; }

extern "C" void kernel_launch(void* const* d_in, const int* in_sizes, int n_in,
                              void* d_out, int out_size, void* d_ws, size_t ws_size,
                              hipStream_t stream) {
  (void)in_sizes; (void)n_in; (void)out_size; (void)ws_size;

  const int*   tokens     = (const int*)d_in[0];
  const int*   parent_idx = (const int*)d_in[1];
  const int*   batch_idx  = (const int*)d_in[2];
  const float* emb        = (const float*)d_in[3];
  const float* wc_w       = (const float*)d_in[4];
  const float* wc_b       = (const float*)d_in[5];
  const float* g0f_wih = (const float*)d_in[6];
  const float* g0f_whh = (const float*)d_in[7];
  const float* g0f_bih = (const float*)d_in[8];
  const float* g0f_bhh = (const float*)d_in[9];
  const float* g0b_wih = (const float*)d_in[10];
  const float* g0b_whh = (const float*)d_in[11];
  const float* g0b_bih = (const float*)d_in[12];
  const float* g0b_bhh = (const float*)d_in[13];
  const float* g1f_wih = (const float*)d_in[14];
  const float* g1f_whh = (const float*)d_in[15];
  const float* g1f_bih = (const float*)d_in[16];
  const float* g1f_bhh = (const float*)d_in[17];
  const float* g1b_wih = (const float*)d_in[18];
  const float* g1b_whh = (const float*)d_in[19];
  const float* g1b_bih = (const float*)d_in[20];
  const float* g1b_bhh = (const float*)d_in[21];
  float* out = (float*)d_out;

  static const int  LS[6]  = {8192, 16384, 32768, 65536, 65536, 65536};
  static const long OFF[7] = {0, 8192, 24576, 57344, 122880, 188416, 245760};

  // ---- bump allocator over d_ws (~160 MB total needed) --------------------
  char* w = (char*)d_ws;
  size_t off = 0;
  auto alloc = [&](size_t bytes) -> void* {
    void* p = w + off;
    off = (off + bytes + 255) & ~(size_t)255;
    return p;
  };

  _Float16* wcw_h   = (_Float16*)alloc((size_t)ENC * ENC * 2);
  _Float16* wih0f_h = (_Float16*)alloc((size_t)GATE3 * ENC * 2);
  _Float16* wih0b_h = (_Float16*)alloc((size_t)GATE3 * ENC * 2);
  _Float16* wih1f_h = (_Float16*)alloc((size_t)GATE3 * 2 * HID * 2);
  _Float16* wih1b_h = (_Float16*)alloc((size_t)GATE3 * 2 * HID * 2);
  _Float16* whh0f_h = (_Float16*)alloc((size_t)GATE3 * HID * 2);
  _Float16* whh0b_h = (_Float16*)alloc((size_t)GATE3 * HID * 2);
  _Float16* whh1f_h = (_Float16*)alloc((size_t)GATE3 * HID * 2);
  _Float16* whh1b_h = (_Float16*)alloc((size_t)GATE3 * HID * 2);
  float*    roots   = (float*)alloc((size_t)RROOTS * ENC * 4);
  _Float16* x0_h    = (_Float16*)alloc((size_t)RROOTS * ENC * 2);
  float*    y0      = (float*)alloc((size_t)RROOTS * 2 * HID * 4);
  _Float16* y0_h    = (_Float16*)alloc((size_t)RROOTS * 2 * HID * 2);
  float*    y1      = (float*)alloc((size_t)RROOTS * 2 * HID * 4);

  // Overlap region: tree-phase buffers, later reused for the gi projections.
  const size_t S = off;
  _Float16* emb_h = (_Float16*)alloc((size_t)VOCABN * ENC * 2);
  float*    P     = (float*)alloc((size_t)VOCABN * ENC * 4);
  float*    hA    = (float*)alloc((size_t)65536 * ENC * 4);
  float*    hB    = (float*)alloc((size_t)65536 * ENC * 4);
  float* giFwd = (float*)(w + S);                    // aliases emb_h/P/hA (dead)
  float* giBwd = giFwd + (size_t)RROOTS * GATE3;

  // Allow >64KB dynamic LDS for the scan kernel (CDNA5: 320KB/workgroup).
  (void)hipFuncSetAttribute((const void*)k_gru_scan,
                            hipFuncAttributeMaxDynamicSharedMemorySize,
                            SCAN_LDS_BYTES);

  // ---- 1) init + f16 weight conversions ----------------------------------
  k_zero_f32<<<grid1d((long)RROOTS * ENC, 256), 256, 0, stream>>>(roots, RROOTS * ENC);
  k_f32_to_f16<<<grid1d(ENC * ENC, 256), 256, 0, stream>>>(wc_w, wcw_h, ENC * ENC);
  k_f32_to_f16<<<grid1d(GATE3 * ENC, 256), 256, 0, stream>>>(g0f_wih, wih0f_h, GATE3 * ENC);
  k_f32_to_f16<<<grid1d(GATE3 * ENC, 256), 256, 0, stream>>>(g0b_wih, wih0b_h, GATE3 * ENC);
  k_f32_to_f16<<<grid1d(GATE3 * 2 * HID, 256), 256, 0, stream>>>(g1f_wih, wih1f_h, GATE3 * 2 * HID);
  k_f32_to_f16<<<grid1d(GATE3 * 2 * HID, 256), 256, 0, stream>>>(g1b_wih, wih1b_h, GATE3 * 2 * HID);
  k_f32_to_f16<<<grid1d(GATE3 * HID, 256), 256, 0, stream>>>(g0f_whh, whh0f_h, GATE3 * HID);
  k_f32_to_f16<<<grid1d(GATE3 * HID, 256), 256, 0, stream>>>(g0b_whh, whh0b_h, GATE3 * HID);
  k_f32_to_f16<<<grid1d(GATE3 * HID, 256), 256, 0, stream>>>(g1f_whh, whh1f_h, GATE3 * HID);
  k_f32_to_f16<<<grid1d(GATE3 * HID, 256), 256, 0, stream>>>(g1b_whh, whh1b_h, GATE3 * HID);
  k_f32_to_f16<<<grid1d((long)VOCABN * ENC, 256), 256, 0, stream>>>(emb, emb_h, VOCABN * ENC);

  // ---- 2) P = emb @ wc_w^T + wc_b   (50000x128x128, WMMA) -----------------
  k_gemm_nt<<<gemm_grid(VOCABN, ENC), 128, 0, stream>>>(emb_h, wcw_h, wc_b, P,
                                                        VOCABN, ENC, ENC);

  // ---- 3) tree aggregation bottom-up + fused segment_max/relu -------------
  float* bufs[2] = {hA, hB};
  for (int l = 5; l >= 0; --l) {
    float* cur = bufs[l & 1];
    const long n = (long)LS[l] * ENC;
    k_gather_proj<<<grid1d(n, 256), 256, 0, stream>>>(P, tokens + OFF[l], cur, LS[l]);
    if (l < 5) {
      float* prev = bufs[(l + 1) & 1];
      const long nc = (long)LS[l + 1] * ENC;
      k_scatter_add<<<grid1d(nc, 256), 256, 0, stream>>>(cur, prev,
                                                         parent_idx + OFF[l + 1], LS[l + 1]);
    }
    k_pool_max<<<grid1d(n, 256), 256, 0, stream>>>(cur, batch_idx + OFF[l],
                                                   (int*)roots, LS[l]);
  }

  // ---- 4) BiGRU layer 0 ---------------------------------------------------
  k_f32_to_f16<<<grid1d((long)RROOTS * ENC, 256), 256, 0, stream>>>(roots, x0_h, RROOTS * ENC);
  k_gemm_nt<<<gemm_grid(RROOTS, GATE3), 128, 0, stream>>>(x0_h, wih0f_h, g0f_bih, giFwd,
                                                          RROOTS, GATE3, ENC);
  k_gemm_nt<<<gemm_grid(RROOTS, GATE3), 128, 0, stream>>>(x0_h, wih0b_h, g0b_bih, giBwd,
                                                          RROOTS, GATE3, ENC);
  k_gru_scan<<<2, 1024, SCAN_LDS_BYTES, stream>>>(giFwd, giBwd, whh0f_h, whh0b_h,
                                                  g0f_bhh, g0b_bhh, y0);

  // ---- 5) BiGRU layer 1 ---------------------------------------------------
  k_f32_to_f16<<<grid1d((long)RROOTS * 2 * HID, 256), 256, 0, stream>>>(y0, y0_h,
                                                                        RROOTS * 2 * HID);
  k_gemm_nt<<<gemm_grid(RROOTS, GATE3), 128, 0, stream>>>(y0_h, wih1f_h, g1f_bih, giFwd,
                                                          RROOTS, GATE3, 2 * HID);
  k_gemm_nt<<<gemm_grid(RROOTS, GATE3), 128, 0, stream>>>(y0_h, wih1b_h, g1b_bih, giBwd,
                                                          RROOTS, GATE3, 2 * HID);
  k_gru_scan<<<2, 1024, SCAN_LDS_BYTES, stream>>>(giFwd, giBwd, whh1f_h, whh1b_h,
                                                  g1f_bhh, g1b_bhh, y1);

  // ---- 6) max over time ---------------------------------------------------
  k_time_max<<<grid1d(BATCH * 2 * HID, 256), 256, 0, stream>>>(y1, out);
}
